// IPI_26886495273226
// MI455X (gfx1250) — matrix operand order repaired and checked by reading, not compiled, not run on hardware
//
#include <hip/hip_runtime.h>
#include <math.h>

// ---------------- problem constants ----------------
constexpr int   DH_    = 50;
constexpr int   DW_    = 50;
constexpr int   STEP_  = 10;
constexpr int   NI_    = 16;     // (200-50)/10+1
constexpr int   NJ_    = 16;
constexpr int   L_     = 256;    // patches (columns)
constexpr int   P_     = 2500;   // pixels per patch (rows)
constexpr int   P2_    = 2560;   // rows padded to multiple of 32
constexpr int   H_     = 200;
constexpr int   W_     = 200;
constexpr int   MAXIT_ = 1000;
constexpr float LAM_   = 0.02f;  // 1/sqrt(2500)
constexpr float TOL_   = 1e-6f;

struct ApgState {
  float tkm1, tk, mu, mu_bar, coef, thr;
  float sumS, sumX, outmax;
  int   cnt, prank, pcard, it, done;
  int   rank_acc, card_acc;
};

typedef __attribute__((ext_vector_type(2)))  float  v2f;
typedef __attribute__((ext_vector_type(8)))  float  v8f;
typedef __attribute__((ext_vector_type(16))) __bf16 v16bf;

#if __has_builtin(__builtin_amdgcn_wmma_f32_16x16x4_f32)
#define HAVE_WMMA_F32 1
#else
#define HAVE_WMMA_F32 0
#endif

__device__ __forceinline__ int imin(int a, int b) { return a < b ? a : b; }
__device__ __forceinline__ int imax(int a, int b) { return a > b ? a : b; }

// ---------------- WMMA tile helpers (wave32, one 16x16 C tile per wave) ----
// element(M,k) of A-tile = A[M*asr + k*asc] ; element(k,N) of B-tile = B[k*bsr + N*bsc]
__device__ __forceinline__ v8f wmma_block(const float* __restrict__ A, int asr, int asc,
                                          const float* __restrict__ B, int bsr, int bsc,
                                          int K) {
  const int lane = threadIdx.x & 31;
  const int m    = lane & 15;
  const int hi   = lane >> 4;
  v8f acc = {0.f, 0.f, 0.f, 0.f, 0.f, 0.f, 0.f, 0.f};
#if HAVE_WMMA_F32
  // fp32 WMMA: A tile 16x4, per-lane v2f holds K = 2*hi, 2*hi+1
  for (int k = 0; k < K; k += 4) {
    const int ka = k + 2 * hi;
    v2f a, b;
    a.x = A[m * asr + ka * asc];
    a.y = A[m * asr + (ka + 1) * asc];
    b.x = B[ka * bsr + m * bsc];
    b.y = B[(ka + 1) * bsr + m * bsc];
    acc = __builtin_amdgcn_wmma_f32_16x16x4_f32(false, a, false, b, (short)0, acc,
                                                false, false);
  }
#else
  // bf16 fallback (codegen-confirmed builtin): A 16x32, B 32x16 per step
  for (int k = 0; k < K; k += 32) {
    v16bf a, b;
#pragma unroll
    for (int e = 0; e < 16; ++e) {
      const int ka = k + ((e < 8) ? (8 * hi + e) : (16 + 8 * hi + (e - 8)));
      const int kb = k + ((e < 8) ? (8 * hi + e) : (16 + 8 * hi + (e - 8)));
      a[e] = (__bf16)A[m * asr + ka * asc];
      b[e] = (__bf16)B[kb * bsr + m * bsc];
    }
    acc = __builtin_amdgcn_wmma_f32_16x16x32_bf16(false, a, false, b, (short)0, acc,
                                                  false, false);
  }
#endif
  return acc;
}

__device__ __forceinline__ void wmma_store(float* __restrict__ C, int ldc, v8f acc) {
  const int lane = threadIdx.x & 31;
  const int m    = lane & 15;
  const int hi   = lane >> 4;
#pragma unroll
  for (int r = 0; r < 8; ++r)
    C[(r + 8 * hi) * ldc + m] = acc[r];
}

// ---------------- setup kernels ----------------
__global__ void k_zero(float* __restrict__ p) {
  p[blockIdx.x * 256 + threadIdx.x] = 0.f;
}

__global__ void k_colminmax(const float* __restrict__ inp,
                            float* __restrict__ cmin, float* __restrict__ cmax) {
  const int l  = blockIdx.x;          // patch id = pi*16+pj
  const int pi = l >> 4, pj = l & 15;
  const int y0 = pi * STEP_, x0 = pj * STEP_;
  const int t  = threadIdx.x;
  float mn = 3.4e38f, mx = -3.4e38f;
  for (int r = t; r < P_; r += 256) {
    const int dy = r / DW_, dx = r % DW_;
    const float v = inp[(y0 + dy) * W_ + (x0 + dx)] * 255.f;
    mn = fminf(mn, v);
    mx = fmaxf(mx, v);
  }
  __shared__ float smn[256], smx[256];
  smn[t] = mn; smx[t] = mx;
  __syncthreads();
  for (int o = 128; o > 0; o >>= 1) {
    if (t < o) { smn[t] = fminf(smn[t], smn[t + o]); smx[t] = fmaxf(smx[t], smx[t + o]); }
    __syncthreads();
  }
  if (t == 0) { cmin[l] = smn[0]; cmax[l] = smx[0]; }
}

__global__ void k_buildD(const float* __restrict__ inp,
                         const float* __restrict__ cmin, const float* __restrict__ cmax,
                         float* __restrict__ Dm) {
  const int i = blockIdx.x * 256 + threadIdx.x;   // i = r*L + l
  const int r = i >> 8, l = i & 255;
  float v = 0.f;
  if (r < P_) {
    const int pi = l >> 4, pj = l & 15;
    const int dy = r / DW_, dx = r % DW_;
    const float x = inp[(pi * STEP_ + dy) * W_ + (pj * STEP_ + dx)] * 255.f;
    v = (x - cmin[l]) / (cmax[l] - cmin[l] + 1e-8f);
  }
  Dm[i] = v;
}

// ---------------- WMMA GEMM kernels ----------------
// G = A^T * A   (A: P2_ x L_, G: 256x256). 256 waves, K = P2_.
__global__ void k_gram(const float* __restrict__ A, float* __restrict__ G,
                       const ApgState* st, int checkDone) {
  if (checkDone && st->done) return;
  const int wid = blockIdx.x * 8 + (threadIdx.x >> 5);
  const int ti = wid >> 4, tj = wid & 15;
  v8f acc = wmma_block(A + ti * 16, /*asr=*/1, /*asc=*/L_,
                       A + tj * 16, /*bsr=*/L_, /*bsc=*/1, P2_);
  wmma_store(G + (ti * 16) * 256 + tj * 16, 256, acc);
}

// C = A * B^T   (A,B,C: 256x256). 256 waves, K = 256.
__global__ void k_gemm_nt(const float* __restrict__ A, const float* __restrict__ B,
                          float* __restrict__ C, const ApgState* st) {
  if (st->done) return;
  const int wid = blockIdx.x * 8 + (threadIdx.x >> 5);
  const int ti = wid >> 4, tj = wid & 15;
  v8f acc = wmma_block(A + (ti * 16) * 256, 256, 1,
                       B + (tj * 16) * 256, /*bsr=*/1, /*bsc=*/256, 256);
  wmma_store(C + (ti * 16) * 256 + tj * 16, 256, acc);
}

// C = A * B     (A: P2_ x 256, B: 256x256, C: P2_ x 256). 2560 waves, K = 256.
__global__ void k_gemm_nn(const float* __restrict__ A, const float* __restrict__ B,
                          float* __restrict__ C, const ApgState* st) {
  if (st->done) return;
  const int wid = blockIdx.x * 8 + (threadIdx.x >> 5);
  const int ti = wid >> 4, tj = wid & 15;
  v8f acc = wmma_block(A + (ti * 16) * L_, L_, 1,
                       B + tj * 16, 256, 1, 256);
  wmma_store(C + (ti * 16) * L_ + tj * 16, L_, acc);
}

// ---------------- parallel block-Jacobi eigensolver (single workgroup) -----
// Diagonalizes G (256x256, global), accumulates V (global). Eigenvalues end on diag(G).
__global__ void k_jacobi(float* __restrict__ G, float* __restrict__ V,
                         const ApgState* st, int checkDone, int nsweeps) {
  if (checkDone && st->done) return;
  __shared__ int   perm[256];
  __shared__ float cs[128], sn[128];
  __shared__ int   pp[128], qq[128];
  const int t = threadIdx.x;
  for (int i = t; i < 256 * 256; i += 256)
    V[i] = ((i >> 8) == (i & 255)) ? 1.f : 0.f;
  perm[t] = t;
  __syncthreads();
  for (int sweep = 0; sweep < nsweeps; ++sweep) {
    for (int round = 0; round < 255; ++round) {
      if (t < 128) {
        int p = perm[t], q = perm[255 - t];
        if (p > q) { int tmp = p; p = q; q = tmp; }
        pp[t] = p; qq[t] = q;
        const float apq = G[p * 256 + q];
        const float app = G[p * 256 + p];
        const float aqq = G[q * 256 + q];
        float c = 1.f, s = 0.f;
        if (fabsf(apq) > 1e-20f) {
          const float th = (aqq - app) / (2.f * apq);
          const float tt = (th >= 0.f ? 1.f : -1.f) / (fabsf(th) + sqrtf(1.f + th * th));
          c = rsqrtf(1.f + tt * tt);
          s = tt * c;
        }
        cs[t] = c; sn[t] = s;
      }
      __syncthreads();
      // columns: G <- G * J (disjoint pairs -> independent)
      for (int w = t; w < 128 * 256; w += 256) {
        const int k = w >> 8, r = w & 255;
        const float c = cs[k], s = sn[k];
        const int p = pp[k], q = qq[k];
        const float gp = G[r * 256 + p], gq = G[r * 256 + q];
        G[r * 256 + p] = c * gp - s * gq;
        G[r * 256 + q] = s * gp + c * gq;
      }
      __syncthreads();
      // rows: G <- J^T * G
      for (int w = t; w < 128 * 256; w += 256) {
        const int k = w >> 8, cc = w & 255;
        const float c = cs[k], s = sn[k];
        const int p = pp[k], q = qq[k];
        const float gp = G[p * 256 + cc], gq = G[q * 256 + cc];
        G[p * 256 + cc] = c * gp - s * gq;
        G[q * 256 + cc] = s * gp + c * gq;
      }
      __syncthreads();
      // V <- V * J
      for (int w = t; w < 128 * 256; w += 256) {
        const int k = w >> 8, r = w & 255;
        const float c = cs[k], s = sn[k];
        const int p = pp[k], q = qq[k];
        const float vp = V[r * 256 + p], vq = V[r * 256 + q];
        V[r * 256 + p] = c * vp - s * vq;
        V[r * 256 + q] = s * vp + c * vq;
      }
      __syncthreads();
      if (t == 0) {                      // round-robin rotate (keep perm[0])
        const int last = perm[255];
        for (int i = 255; i > 1; --i) perm[i] = perm[i - 1];
        perm[1] = last;
      }
      __syncthreads();
    }
  }
}

// ---------------- init of APG scalar state from S0 -------------------------
__global__ void k_mu0(const float* __restrict__ G, ApgState* st) {
  __shared__ float s[256];
  const int t = threadIdx.x;
  s[t] = sqrtf(fmaxf(G[t * 256 + t], 0.f));
  __syncthreads();
  for (int phase = 0; phase < 256; ++phase) {      // even-odd sort, descending
    const int i = 2 * t + (phase & 1);
    if (i + 1 < 256) {
      const float a = s[i], b = s[i + 1];
      if (a < b) { s[i] = b; s[i + 1] = a; }
    }
    __syncthreads();
  }
  if (t == 0) {
    st->mu     = s[1];
    st->mu_bar = 0.005f * s[3];
    st->tkm1 = 1.f; st->tk = 1.f;
    st->cnt = 0; st->prank = 0; st->pcard = 0; st->it = 0; st->done = 0;
    st->outmax = -3.4e38f;
  }
}

// ---------------- per-iteration kernels ----------------
__global__ void k_pre(ApgState* st) {
  if (st->done) return;
  st->coef = (st->tkm1 - 1.f) / st->tk;
  st->thr  = st->mu * 0.5f;              // mu / TAU
  st->sumS = 0.f; st->sumX = 0.f;
  st->card_acc = 0; st->rank_acc = 0;
}

__global__ void k_ew1(const float* __restrict__ Dm,
                      const float* __restrict__ Xm1A, const float* __restrict__ XA,
                      const float* __restrict__ Xm1E, const float* __restrict__ XE,
                      float* __restrict__ YA, float* __restrict__ YE,
                      float* __restrict__ GA, float* __restrict__ GE,
                      const ApgState* st) {
  if (st->done) return;
  const int i = blockIdx.x * 256 + threadIdx.x;
  const float coef = st->coef;
  const float xa = XA[i], xe = XE[i];
  const float ya = xa + coef * (xa - Xm1A[i]);
  const float ye = xe + coef * (xe - Xm1E[i]);
  const float tv = (ya + ye - Dm[i]) * 0.5f;       // T = (YA+YE-D)/TAU
  YA[i] = ya; YE[i] = ye;
  GA[i] = ya - tv; GE[i] = ye - tv;
}

__global__ void k_specthr(const float* __restrict__ G, float* __restrict__ f,
                          ApgState* st) {
  if (st->done) return;
  __shared__ int cnt[256];
  const int t = threadIdx.x;
  const float sv  = sqrtf(fmaxf(G[t * 256 + t], 0.f));
  const float thr = st->thr;
  const int active = sv > thr;
  f[t] = active ? (sv - thr) / sv : 0.f;
  cnt[t] = active;
  __syncthreads();
  for (int o = 128; o > 0; o >>= 1) { if (t < o) cnt[t] += cnt[t + o]; __syncthreads(); }
  if (t == 0) st->rank_acc = cnt[0];
}

__global__ void k_scaleV(const float* __restrict__ V, const float* __restrict__ f,
                         float* __restrict__ Vf, const ApgState* st) {
  if (st->done) return;
  const int i = blockIdx.x * 256 + threadIdx.x;    // i = r*256 + k
  Vf[i] = V[i] * f[i & 255];
}

__global__ void k_ew2(const float* __restrict__ GE, const float* __restrict__ YA,
                      const float* __restrict__ YE, const float* __restrict__ X1A,
                      float* __restrict__ X1E, ApgState* st) {
  if (st->done) return;
  const int i = blockIdx.x * 256 + threadIdx.x;
  const float lthr = LAM_ * st->thr;
  const float ge  = GE[i];
  const float mag = fabsf(ge) - lthr;
  const float x1e = (mag > 0.f) ? ((ge > 0.f) ? mag : -mag) : 0.f;
  X1E[i] = x1e;
  const float x1a = X1A[i], ya = YA[i], ye = YE[i];
  const float tmp = x1a + x1e - ya - ye;
  const float sa = 2.f * (ya - x1a) + tmp;         // TAU*(YA-X1A)+tmp
  const float se = 2.f * (ye - x1e) + tmp;
  float ls = sa * sa + se * se;
  float lx = x1a * x1a + x1e * x1e;
  int   lc = (x1e != 0.f);
  __shared__ float rs[256], rx[256];
  __shared__ int   rc[256];
  const int t = threadIdx.x;
  rs[t] = ls; rx[t] = lx; rc[t] = lc;
  __syncthreads();
  for (int o = 128; o > 0; o >>= 1) {
    if (t < o) { rs[t] += rs[t + o]; rx[t] += rx[t + o]; rc[t] += rc[t + o]; }
    __syncthreads();
  }
  if (t == 0) {
    atomicAdd(&st->sumS, rs[0]);
    atomicAdd(&st->sumX, rx[0]);
    atomicAdd(&st->card_acc, rc[0]);
  }
}

__global__ void k_commit(float* __restrict__ Xm1A, float* __restrict__ XA,
                         const float* __restrict__ X1A,
                         float* __restrict__ Xm1E, float* __restrict__ XE,
                         const float* __restrict__ X1E, const ApgState* st) {
  if (st->done) return;
  const int i = blockIdx.x * 256 + threadIdx.x;
  const float xa = XA[i], xe = XE[i];
  Xm1A[i] = xa; XA[i] = X1A[i];
  Xm1E[i] = xe; XE[i] = X1E[i];
}

__global__ void k_post(ApgState* st) {
  if (st->done) return;
  const float num = sqrtf(st->sumS);
  const float den = 2.0f * fmaxf(1.f, sqrtf(st->sumX));
  int conv = (num / den) <= TOL_;
  const int rank = st->rank_acc, card = st->card_acc;
  const float mu1 = fmaxf(0.9f * st->mu, st->mu_bar);
  const int same = (rank == st->prank);
  const int cnt1 = same ? st->cnt + 1 : 0;
  int dc = card - st->pcard; if (dc < 0) dc = -dc;
  if (same && (cnt1 > 10) && (dc < 20)) conv = 1;
  const int pcard1 = same ? st->pcard : card;
  if (rank > 76) conv = 1;                         // rank_cap = 0.3*256 = 76.8
  if (st->it + 1 >= MAXIT_) conv = 1;
  const float t1 = 0.5f * (1.f + sqrtf(1.f + 4.f * st->tk * st->tk));
  st->tkm1 = st->tk; st->tk = t1;
  st->mu = mu1; st->cnt = cnt1; st->prank = rank; st->pcard = pcard1;
  st->it += 1; st->done = conv;
}

// ---------------- median aggregation + normalization ----------------
__device__ __forceinline__ int fdiv(int a, int b) {
  int q = a / b;
  if ((a % b) && ((a ^ b) < 0)) --q;
  return q;
}

__device__ __forceinline__ void atomicMaxF(float* addr, float val) {
  unsigned int* ua = (unsigned int*)addr;
  unsigned int old = *ua;
  while (true) {
    const float f = __uint_as_float(old);
    if (f >= val) break;
    const unsigned int assumed = old;
    old = atomicCAS(ua, assumed, __float_as_uint(val));
    if (old == assumed) break;
  }
}

__global__ void k_median(const float* __restrict__ XE, float* __restrict__ Mout,
                         ApgState* st) {
  const int i = blockIdx.x * 256 + threadIdx.x;
  if (i >= H_ * W_) return;
  const int y = i / W_, x = i % W_;
  const int pi_lo = imax(0, -fdiv(DH_ - 1 - y, STEP_));
  const int hi_i  = imin(NI_ - 1, y / STEP_);
  const int nrow  = imax(0, hi_i - pi_lo + 1);
  const int pj_lo = imax(0, -fdiv(DW_ - 1 - x, STEP_));
  const int hi_j  = imin(NJ_ - 1, x / STEP_);
  const int ncol  = imax(0, hi_j - pj_lo + 1);
  float vals[25];
#pragma unroll
  for (int a = 0; a < 5; ++a) {
    const int pi = imin(imax(pi_lo + a, 0), NI_ - 1);
    const int ii = imin(imax(y - pi * STEP_, 0), DH_ - 1);
#pragma unroll
    for (int b = 0; b < 5; ++b) {
      const int pj = imin(imax(pj_lo + b, 0), NJ_ - 1);
      const int jj = imin(imax(x - pj * STEP_, 0), DW_ - 1);
      float v = 3.4e38f;                                     // +inf pad
      if (a < nrow && b < ncol)
        v = XE[(ii * DW_ + jj) * L_ + (pi * NJ_ + pj)];
      vals[a * 5 + b] = v;
    }
  }
  for (int a = 1; a < 25; ++a) {                              // insertion sort
    const float key = vals[a];
    int b = a - 1;
    while (b >= 0 && vals[b] > key) { vals[b + 1] = vals[b]; --b; }
    vals[b + 1] = key;
  }
  const int C = nrow * ncol;
  float med = 0.f;
  if (C > 0) med = vals[imin(imax((C - 1) / 2, 0), 24)];
  Mout[i] = med;
  atomicMaxF(&st->outmax, med);
}

__global__ void k_normalize(const float* __restrict__ Mout, const ApgState* st,
                            float* __restrict__ out) {
  const int i = blockIdx.x * 256 + threadIdx.x;
  if (i >= H_ * W_) return;
  out[i] = Mout[i] / (st->outmax + 1e-10f);
}

// ---------------- host driver ----------------
extern "C" void kernel_launch(void* const* d_in, const int* in_sizes, int n_in,
                              void* d_out, int out_size, void* d_ws, size_t ws_size,
                              hipStream_t stream) {
  (void)in_sizes; (void)n_in; (void)out_size; (void)ws_size;
  const float* inp = (const float*)d_in[0];
  float* out = (float*)d_out;

  char* ws = (char*)d_ws;
  size_t off = 0;
  auto alloc = [&](size_t nfloats) -> float* {
    float* p = (float*)(ws + off);
    off = (off + nfloats * sizeof(float) + 255) & ~(size_t)255;
    return p;
  };
  ApgState* st = (ApgState*)ws; off = 256;
  const size_t PL = (size_t)P2_ * L_;          // 655360
  float* Dm   = alloc(PL);
  float* Xm1A = alloc(PL);
  float* XA   = alloc(PL);
  float* X1A  = alloc(PL);
  float* Xm1E = alloc(PL);
  float* XE   = alloc(PL);
  float* X1E  = alloc(PL);
  float* YA   = alloc(PL);
  float* YE   = alloc(PL);
  float* GAb  = alloc(PL);
  float* GEb  = alloc(PL);
  float* G    = alloc(256 * 256);
  float* V    = alloc(256 * 256);
  float* Vf   = alloc(256 * 256);
  float* Wm   = alloc(256 * 256);
  float* fvec = alloc(256);
  float* cmin = alloc(256);
  float* cmax = alloc(256);
  float* Mout = alloc(H_ * W_);

  const int EB = (int)(PL / 256);              // 2560 blocks for elementwise grids
  const int MB = (H_ * W_ + 255) / 256;        // 157 blocks for 200x200 grids

  // ---- build D (unfold + per-patch min/max normalize), zero APG state ----
  k_colminmax<<<L_, 256, 0, stream>>>(inp, cmin, cmax);
  k_buildD<<<EB, 256, 0, stream>>>(inp, cmin, cmax, Dm);
  k_zero<<<EB, 256, 0, stream>>>(Xm1A);
  k_zero<<<EB, 256, 0, stream>>>(XA);
  k_zero<<<EB, 256, 0, stream>>>(Xm1E);
  k_zero<<<EB, 256, 0, stream>>>(XE);

  // ---- initial spectrum: S0 via Gram + Jacobi -> mu0, mu_bar ----
  k_gram<<<32, 256, 0, stream>>>(Dm, G, st, 0);
  k_jacobi<<<1, 256, 0, stream>>>(G, V, st, 0, 10);
  k_mu0<<<1, 256, 0, stream>>>(G, st);

  // ---- APG iterations (fixed launch count; device-side early-out) ----
  for (int it = 0; it < MAXIT_; ++it) {
    k_pre<<<1, 1, 0, stream>>>(st);
    k_ew1<<<EB, 256, 0, stream>>>(Dm, Xm1A, XA, Xm1E, XE, YA, YE, GAb, GEb, st);
    k_gram<<<32, 256, 0, stream>>>(GAb, G, st, 1);
    k_jacobi<<<1, 256, 0, stream>>>(G, V, st, 1, 8);
    k_specthr<<<1, 256, 0, stream>>>(G, fvec, st);
    k_scaleV<<<256, 256, 0, stream>>>(V, fvec, Vf, st);
    k_gemm_nt<<<32, 256, 0, stream>>>(Vf, V, Wm, st);        // Wm = Vf * V^T
    k_gemm_nn<<<320, 256, 0, stream>>>(GAb, Wm, X1A, st);    // X1A = GA * Wm
    k_ew2<<<EB, 256, 0, stream>>>(GEb, YA, YE, X1A, X1E, st);
    k_commit<<<EB, 256, 0, stream>>>(Xm1A, XA, X1A, Xm1E, XE, X1E, st);
    k_post<<<1, 1, 0, stream>>>(st);
  }

  // ---- median aggregation back to 200x200 and max-normalize ----
  k_median<<<MB, 256, 0, stream>>>(XE, Mout, st);
  k_normalize<<<MB, 256, 0, stream>>>(Mout, st, out);
}
